// BindingAffinityGNN_48172353191955
// MI455X (gfx1250) — matrix operand order repaired and checked by reading, not compiled, not run on hardware
//
#include <hip/hip_runtime.h>
#include <math.h>

#define N_NODES 32768
#define B_GRAPHS 512
#define NPG 64
#define EPG 1024
#define HEADS 10
#define CC 20
#define HC 200   // HEADS*CC
#define KTOP 20

typedef __attribute__((ext_vector_type(16))) _Float16 v16h;
typedef __attribute__((ext_vector_type(8)))  float    v8f;

// ---------------------------------------------------------------------------
// WMMA GEMM:  Y[M x Nout] = X[M x 20] @ W[20 x Nout] (+ bias)
// One wave32 per 16x16 output tile; K=20 zero-padded into a single
// v_wmma_f32_16x16x32_f16 (fp32 accumulate).
// A frag (16-bit A 16x32 layout): lane m=lane&15, grp=lane>>4;
//   half index 2v+h holds K = ((v<4)?2v:2v+8) + 8*grp + h
// B frag (32x16): lane n=lane&15; half 2v+h holds K = ((v<4)?2v:2v+8) + 8*grp + h
// D: lane n=lane&15; acc[r] is row r + 8*grp.
// ---------------------------------------------------------------------------
__global__ void __launch_bounds__(256)
gemm_k20_wmma(const float* __restrict__ X,
              const float* __restrict__ W,
              const float* __restrict__ bias,
              float* __restrict__ Y,
              int M, int Nout)
{
  const int K = 20;
  int warp = threadIdx.x >> 5;
  int lane = threadIdx.x & 31;
  int ntn  = (Nout + 15) >> 4;
  int tiles = (M >> 4) * ntn;
  int tile = blockIdx.x * (blockDim.x >> 5) + warp;
  if (tile >= tiles) return;               // warp-uniform exit (EXEC all-1 in wmma)
  int tm = tile / ntn, tn = tile % ntn;
  int grp = lane >> 4, lm = lane & 15;
  int row  = tm * 16 + lm;                 // M is a multiple of 16
  int coln = tn * 16 + lm;
  const float* xrow = X + (size_t)row * K;
  v16h a, b;
#pragma unroll
  for (int v = 0; v < 8; ++v) {
#pragma unroll
    for (int h = 0; h < 2; ++h) {
      int kA = ((v < 4) ? 2 * v : 2 * v + 8) + 8 * grp + h;
      a[2 * v + h] = (kA < K) ? (_Float16)xrow[kA] : (_Float16)0.0f;
      int kB = ((v < 4) ? 2 * v : 2 * v + 8) + 8 * grp + h;
      float wv = (kB < K && coln < Nout) ? W[(size_t)kB * Nout + coln] : 0.0f;
      b[2 * v + h] = (_Float16)wv;
    }
  }
  v8f acc = {};
  acc = __builtin_amdgcn_wmma_f32_16x16x32_f16(false, a, false, b,
                                               (short)0, acc, false, false);
  if (coln < Nout) {
    float bv = bias ? bias[coln] : 0.0f;
#pragma unroll
    for (int r = 0; r < 8; ++r) {
      int m = tm * 16 + r + 8 * grp;
      Y[(size_t)m * Nout + coln] = acc[r] + bv;
    }
  }
}

// ---- monotonic float<->uint encoding so unsigned atomicMax == float max ----
__device__ inline unsigned enc_f(float f) {
  unsigned u = __float_as_uint(f);
  return (u & 0x80000000u) ? ~u : (u | 0x80000000u);
}
__device__ inline float dec_f(unsigned e) {
  unsigned u = (e & 0x80000000u) ? (e ^ 0x80000000u) : ~e;
  return __uint_as_float(u);
}

// ---------------------------------------------------------------------------
// GATv2 layer, one workgroup per graph, all edge work in LDS (~63KB).
// ---------------------------------------------------------------------------
__global__ void __launch_bounds__(256)
gatv2_kernel(const float* __restrict__ xl,     // N x 200 (x @ W_l + b_l)
             const float* __restrict__ xr,     // N x 200 (x @ W_r + b_r)
             const int* __restrict__ src,
             const int* __restrict__ dst,
             const float* __restrict__ eattr,  // E x 8
             const float* __restrict__ W_e,    // 8 x 200
             const float* __restrict__ att,    // 10 x 20
             const float* __restrict__ bias_gat, // 20
             float* __restrict__ out)          // N x 20
{
  const int b = blockIdx.x;
  const int tid = threadIdx.x;
  const int e0 = b * EPG;
  const int n0 = b * NPG;
  const int NE = EPG + NPG;                // real edges + self loops

  __shared__ float    s_logit[NE * HEADS];   // 43520 B
  __shared__ unsigned s_max[NPG * HEADS];    //  2560 B
  __shared__ float    s_sum[NPG * HEADS];    //  2560 B
  __shared__ float    s_lattr[NPG * 8];      //  2048 B
  __shared__ float    s_deg[NPG];            //   256 B
  __shared__ float    s_m[NPG * CC];         //  5120 B
  __shared__ float    s_We[8 * HC];          //  6400 B
  __shared__ float    s_att[HEADS * CC];     //   800 B

  for (int i = tid; i < NPG * HEADS; i += 256) { s_max[i] = 0u; s_sum[i] = 0.0f; }
  for (int i = tid; i < NPG * 8; i += 256) s_lattr[i] = 0.0f;
  for (int i = tid; i < NPG; i += 256) s_deg[i] = 0.0f;
  for (int i = tid; i < NPG * CC; i += 256) s_m[i] = 0.0f;
  for (int i = tid; i < 8 * HC; i += 256) s_We[i] = W_e[i];
  for (int i = tid; i < HEADS * CC; i += 256) s_att[i] = att[i];
  __syncthreads();

  // pass 0: in-degree + mean incoming edge_attr (self-loop fill 'mean')
  for (int e = tid; e < EPG; e += 256) {
    int dl = dst[e0 + e] - n0;
    atomicAdd(&s_deg[dl], 1.0f);
    const float* ea = eattr + (size_t)(e0 + e) * 8;
#pragma unroll
    for (int k = 0; k < 8; ++k) atomicAdd(&s_lattr[dl * 8 + k], ea[k]);
  }
  __syncthreads();
  for (int n = tid; n < NPG; n += 256) {
    float inv = 1.0f / fmaxf(s_deg[n], 1.0f);
#pragma unroll
    for (int k = 0; k < 8; ++k) s_lattr[n * 8 + k] *= inv;
  }
  __syncthreads();

  // pass 1: logits + per-(node,head) segment max
  for (int e = tid; e < NE; e += 256) {
    int sg, dg; float a8[8];
    if (e < EPG) {
      sg = src[e0 + e]; dg = dst[e0 + e];
      const float* ea = eattr + (size_t)(e0 + e) * 8;
#pragma unroll
      for (int k = 0; k < 8; ++k) a8[k] = ea[k];
    } else {
      int n = e - EPG; sg = dg = n0 + n;
#pragma unroll
      for (int k = 0; k < 8; ++k) a8[k] = s_lattr[n * 8 + k];
    }
    int dl = dg - n0;
    const float* XL = xl + (size_t)sg * HC;
    const float* XR = xr + (size_t)dg * HC;
    for (int h = 0; h < HEADS; ++h) {
      float lg = 0.0f;
      for (int c = 0; c < CC; ++c) {
        int ch = h * CC + c;
        float ee = 0.0f;
#pragma unroll
        for (int k = 0; k < 8; ++k) ee = fmaf(a8[k], s_We[k * HC + ch], ee);
        float v = XL[ch] + XR[ch] + ee;
        v = (v > 0.0f) ? v : 0.2f * v;            // leaky_relu 0.2
        lg = fmaf(s_att[ch], v, lg);
      }
      s_logit[e * HEADS + h] = lg;
      atomicMax(&s_max[dl * HEADS + h], enc_f(lg));
    }
  }
  __syncthreads();

  // pass 2: exp(logit - max) + segment sum
  for (int idx = tid; idx < NE * HEADS; idx += 256) {
    int e = idx / HEADS; int h = idx - e * HEADS;
    int dl = (e < EPG) ? (dst[e0 + e] - n0) : (e - EPG);
    float mx = dec_f(s_max[dl * HEADS + h]);
    float a = __expf(s_logit[idx] - mx);
    s_logit[idx] = a;
    atomicAdd(&s_sum[dl * HEADS + h], a);
  }
  __syncthreads();

  // pass 3: alpha-weighted aggregation of xl[src], mean over heads
  for (int e = tid; e < NE; e += 256) {
    int sg, dl;
    if (e < EPG) { sg = src[e0 + e]; dl = dst[e0 + e] - n0; }
    else         { sg = n0 + (e - EPG); dl = e - EPG; }
    const float* XL = xl + (size_t)sg * HC;
    __builtin_prefetch(XL, 0, 3);
    float alpha[HEADS];
#pragma unroll
    for (int h = 0; h < HEADS; ++h)
      alpha[h] = s_logit[e * HEADS + h] / s_sum[dl * HEADS + h];
    for (int c = 0; c < CC; ++c) {
      float acc = 0.0f;
#pragma unroll
      for (int h = 0; h < HEADS; ++h) acc = fmaf(alpha[h], XL[h * CC + c], acc);
      atomicAdd(&s_m[dl * CC + c], acc * (1.0f / HEADS));
    }
  }
  __syncthreads();

  for (int i = tid; i < NPG * CC; i += 256)
    out[(size_t)n0 * CC + i] = s_m[i] + bias_gat[i % CC];
}

// ---------------------------------------------------------------------------
// Per-graph segment_sum: out[dst] += V[src], feature width F (<= 60)
// ---------------------------------------------------------------------------
__global__ void __launch_bounds__(256)
scatter_sum_kernel(const float* __restrict__ V, const int* __restrict__ src,
                   const int* __restrict__ dst, float* __restrict__ out, int F)
{
  const int b = blockIdx.x, tid = threadIdx.x;
  const int e0 = b * EPG, n0 = b * NPG;
  __shared__ float acc[NPG * 60];
  for (int i = tid; i < NPG * F; i += 256) acc[i] = 0.0f;
  __syncthreads();
  for (int e = tid; e < EPG; e += 256) {
    int sg = src[e0 + e];
    int dl = dst[e0 + e] - n0;
    const float* vs = V + (size_t)sg * F;
    for (int f = 0; f < F; ++f) atomicAdd(&acc[dl * F + f], vs[f]);
  }
  __syncthreads();
  for (int i = tid; i < NPG * F; i += 256) out[(size_t)n0 * F + i] = acc[i];
}

// ---------------------------------------------------------------------------
// GRUCell combine (gi/gh are N x 60, split r|z|n)
// ---------------------------------------------------------------------------
__global__ void gru_kernel(const float* __restrict__ x,
                           const float* __restrict__ gi,
                           const float* __restrict__ gh,
                           float* __restrict__ h)
{
  int i = blockIdx.x * blockDim.x + threadIdx.x;
  if (i >= N_NODES * CC) return;
  int n = i / CC, c = i - n * CC;
  float ir = gi[n * 60 + c], iz = gi[n * 60 + 20 + c], in_ = gi[n * 60 + 40 + c];
  float hr = gh[n * 60 + c], hz = gh[n * 60 + 20 + c], hn  = gh[n * 60 + 40 + c];
  float r = 1.0f / (1.0f + __expf(-(ir + hr)));
  float z = 1.0f / (1.0f + __expf(-(iz + hz)));
  float nn = tanhf(in_ + r * hn);
  h[i] = (1.0f - z) * nn + z * x[i];
}

__global__ void concat_kernel(const float* __restrict__ h0,
                              const float* __restrict__ h1,
                              const float* __restrict__ h2,
                              float* __restrict__ H)
{
  int i = blockIdx.x * blockDim.x + threadIdx.x;
  if (i >= N_NODES * 60) return;
  int n = i / 60, j = i - n * 60;
  float v;
  if (j < 20)      v = h0[n * 20 + j];
  else if (j < 40) v = h1[n * 20 + (j - 20)];
  else             v = h2[n * 20 + (j - 40)];
  H[i] = v;
}

__global__ void score_kernel(const float* __restrict__ neigh,
                             const float* __restrict__ H,
                             const float* __restrict__ W_rel,
                             const float* __restrict__ b_rel,
                             const float* __restrict__ W_root,
                             float* __restrict__ score)
{
  int n = blockIdx.x * blockDim.x + threadIdx.x;
  if (n >= N_NODES) return;
  float s = b_rel[0];
  const float* a = neigh + (size_t)n * 60;
  const float* h = H + (size_t)n * 60;
#pragma unroll 4
  for (int k = 0; k < 60; ++k) s += a[k] * W_rel[k] + h[k] * W_root[k];
  score[n] = tanhf(s);
}

// ---------------------------------------------------------------------------
// SAGPool top-K (rank-by-count, ties -> lower index, matches lax.top_k),
// score-scaled add pool, then the 60->40->30->1 leaky MLP head.
// ---------------------------------------------------------------------------
__global__ void __launch_bounds__(64)
pool_mlp_kernel(const float* __restrict__ H, const float* __restrict__ score,
                const float* __restrict__ W1, const float* __restrict__ b1,
                const float* __restrict__ W2, const float* __restrict__ b2,
                const float* __restrict__ Wo, const float* __restrict__ bo,
                float* __restrict__ out)
{
  const int b = blockIdx.x, tid = threadIdx.x;   // 64 threads = NPG
  __shared__ float sc[NPG];
  __shared__ float g[60];
  __shared__ float t1[40];
  __shared__ float t2[30];
  sc[tid] = score[b * NPG + tid];
  if (tid < 60) g[tid] = 0.0f;
  __syncthreads();
  float my = sc[tid];
  int rank = 0;
  for (int j = 0; j < NPG; ++j) {
    float v = sc[j];
    rank += (v > my) || (v == my && j < tid);
  }
  if (rank < KTOP) {
    const float* h = H + (size_t)(b * NPG + tid) * 60;
    for (int f = 0; f < 60; ++f) atomicAdd(&g[f], my * h[f]);
  }
  __syncthreads();
  if (tid < 40) {
    float a = b1[tid];
    for (int k = 0; k < 60; ++k) a = fmaf(g[k], W1[k * 40 + tid], a);
    t1[tid] = (a > 0.0f) ? a : 0.01f * a;
  }
  __syncthreads();
  if (tid < 30) {
    float a = b2[tid];
    for (int k = 0; k < 40; ++k) a = fmaf(t1[k], W2[k * 30 + tid], a);
    t2[tid] = (a > 0.0f) ? a : 0.01f * a;
  }
  __syncthreads();
  if (tid == 0) {
    float a = bo[0];
    for (int k = 0; k < 30; ++k) a = fmaf(t2[k], Wo[k], a);
    out[b] = a;
  }
}

extern "C" void kernel_launch(void* const* d_in, const int* in_sizes, int n_in,
                              void* d_out, int out_size, void* d_ws, size_t ws_size,
                              hipStream_t stream)
{
  (void)n_in; (void)out_size; (void)ws_size;
  const float* x        = (const float*)d_in[0];
  const int*   ei       = (const int*)  d_in[1];
  const float* ea       = (const float*)d_in[2];
  // d_in[3] = batch (implicit in per-graph layout)
  const float* W_l      = (const float*)d_in[4];
  const float* b_l      = (const float*)d_in[5];
  const float* W_r      = (const float*)d_in[6];
  const float* b_r      = (const float*)d_in[7];
  const float* W_e      = (const float*)d_in[8];
  const float* att      = (const float*)d_in[9];
  const float* bias_gat = (const float*)d_in[10];
  const float* Wg       = (const float*)d_in[11];
  const float* W_ih     = (const float*)d_in[12];
  const float* b_ih     = (const float*)d_in[13];
  const float* W_hh     = (const float*)d_in[14];
  const float* b_hh     = (const float*)d_in[15];
  const float* W_rel    = (const float*)d_in[16];
  const float* b_rel    = (const float*)d_in[17];
  const float* W_root   = (const float*)d_in[18];
  const float* W1       = (const float*)d_in[19];
  const float* b1       = (const float*)d_in[20];
  const float* W2       = (const float*)d_in[21];
  const float* b2       = (const float*)d_in[22];
  const float* Wo       = (const float*)d_in[23];
  const float* bo       = (const float*)d_in[24];

  const int E = in_sizes[1] / 2;
  const int* src = ei;
  const int* dst = ei + E;

  float* ws = (float*)d_ws;
  float* xl = ws;  ws += (size_t)N_NODES * HC;
  float* xr = ws;  ws += (size_t)N_NODES * HC;
  float* p1 = ws;  ws += (size_t)N_NODES * CC;   // m1 / m2
  float* p2 = ws;  ws += (size_t)N_NODES * CC;   // mg / h2
  float* p3 = ws;  ws += (size_t)N_NODES * CC;   // agg
  float* p4 = ws;  ws += (size_t)N_NODES * CC;   // h1
  float* c1 = ws;  ws += (size_t)N_NODES * 60;   // gi / H
  float* c2 = ws;  ws += (size_t)N_NODES * 60;   // gh / neigh
  float* sc = ws;  ws += (size_t)N_NODES;        // scores

  auto gemm = [&](const float* X, const float* W, const float* bias,
                  float* Y, int Nout) {
    int ntn = (Nout + 15) >> 4;
    int tiles = (N_NODES >> 4) * ntn;
    int blocks = (tiles + 7) / 8;                 // 8 waves / block
    gemm_k20_wmma<<<blocks, 256, 0, stream>>>(X, W, bias, Y, N_NODES, Nout);
  };

  // ---- layer 1: GATv2 on x ----
  gemm(x, W_l, b_l, xl, HC);
  gemm(x, W_r, b_r, xr, HC);
  gatv2_kernel<<<B_GRAPHS, 256, 0, stream>>>(xl, xr, src, dst, ea, W_e, att, bias_gat, p1);
  // ---- GatedGraphConv 1 ----
  gemm(p1, Wg, nullptr, p2, CC);
  scatter_sum_kernel<<<B_GRAPHS, 256, 0, stream>>>(p2, src, dst, p3, CC);
  gemm(p3, W_ih, b_ih, c1, 60);
  gemm(p1, W_hh, b_hh, c2, 60);
  gru_kernel<<<(N_NODES * CC + 255) / 256, 256, 0, stream>>>(p1, c1, c2, p4);  // h1
  // ---- layer 2: GATv2 on h1 ----
  gemm(p4, W_l, b_l, xl, HC);
  gemm(p4, W_r, b_r, xr, HC);
  gatv2_kernel<<<B_GRAPHS, 256, 0, stream>>>(xl, xr, src, dst, ea, W_e, att, bias_gat, p1);
  // ---- GatedGraphConv 2 ----
  gemm(p1, Wg, nullptr, p2, CC);
  scatter_sum_kernel<<<B_GRAPHS, 256, 0, stream>>>(p2, src, dst, p3, CC);
  gemm(p3, W_ih, b_ih, c1, 60);
  gemm(p1, W_hh, b_hh, c2, 60);
  gru_kernel<<<(N_NODES * CC + 255) / 256, 256, 0, stream>>>(p1, c1, c2, p2);  // h2
  // ---- H = [x | h1 | h2], SAGPool score, pool + MLP ----
  concat_kernel<<<(N_NODES * 60 + 255) / 256, 256, 0, stream>>>(x, p4, p2, c1);
  scatter_sum_kernel<<<B_GRAPHS, 256, 0, stream>>>(c1, src, dst, c2, 60);
  score_kernel<<<(N_NODES + 255) / 256, 256, 0, stream>>>(c2, c1, W_rel, b_rel, W_root, sc);
  pool_mlp_kernel<<<B_GRAPHS, 64, 0, stream>>>(c1, sc, W1, b1, W2, b2, Wo, bo, (float*)d_out);
}
